// Net_56049323213278
// MI455X (gfx1250) — matrix operand order repaired and checked by reading, not compiled, not run on hardware
//
#include <hip/hip_runtime.h>
#include <hip/hip_bf16.h>
#include <math.h>

typedef __attribute__((ext_vector_type(16))) __bf16 v16bf;
typedef __attribute__((ext_vector_type(8)))  __bf16 v8bf;
typedef __attribute__((ext_vector_type(8)))  float  v8f;

// ---------------------------------------------------------------------------
// Degree / normalization kernels
// ---------------------------------------------------------------------------
__global__ void k_deg_init(float* deg, int n) {
    int i = blockIdx.x * blockDim.x + threadIdx.x;
    if (i < n) deg[i] = 1.0f;                 // self loop contributes 1
}

__global__ void k_deg_count(const long long* __restrict__ ei, float* deg, int E) {
    int e = blockIdx.x * blockDim.x + threadIdx.x;
    if (e < E) {
        int c = (int)ei[(size_t)E + e];       // col = target index
        atomicAdd(&deg[c], 1.0f);
    }
}

__global__ void k_rsqrt_inplace(float* deg, int n) {
    int i = blockIdx.x * blockDim.x + threadIdx.x;
    if (i < n) {
        float d = deg[i];
        deg[i] = (d > 0.0f) ? rsqrtf(d) : 0.0f;
    }
}

__global__ void k_zero(float* p, int n) {
    int i = blockIdx.x * blockDim.x + threadIdx.x;
    if (i < n) p[i] = 0.0f;
}

// ---------------------------------------------------------------------------
// Tiled bf16 WMMA GEMM:  Y[M,F] = X[M,K] @ W[K,F]   (fp32 in/out, bf16 MACs)
//
// Block: 128 threads = 4 waves; BM = 64 (16 rows per wave), BK = 32.
// Double-buffered LDS, software-pipelined: global loads for tile i+1 are
// issued before the WMMAs on tile i, then converted/stored, one barrier/iter.
//
// LDS layouts chosen so fragments are contiguous vector loads:
//   sx[buf][row][40]      : 80 B row stride (16B-aligned, 20-bank stride
//                           -> 16 rows hit disjoint 4-bank groups)
//   swf[buf][t][lane][16] : B-fragment order; lane's 16 bf16 = 32 contiguous
//                           bytes, written as ONE v16bf store and read as ONE
//                           v16bf load (2x ds_*_b128 each)
// ---------------------------------------------------------------------------
template<int F>
__global__ void __launch_bounds__(128)
gcn_gemm_bf16(const float* __restrict__ X, const float* __restrict__ W,
              float* __restrict__ Y, int M, int K) {
    constexpr int BM = 64;
    constexpr int BK = 32;
    constexpr int NT = F / 16;                // 16x16 col tiles per wave
    constexpr int XR = (BM * BK / 4) / 128;   // float4 X loads per thread = 4
    constexpr int WSLOTS = NT * 32;           // fragment lane-slots (<=128)

    __shared__ alignas(16) __bf16 sx[2][BM][40];
    __shared__ alignas(32) __bf16 swf[2][NT][32][16];

    const int tid  = threadIdx.x;
    const int lane = tid & 31;
    const int wave = tid >> 5;
    const int hi   = lane >> 4;         // half-wave selector
    const int lo   = lane & 15;
    const int rowbase = blockIdx.x * BM;
    const int nIter = K / BK;

    // W staging slot for this thread (column-wise fragment gather)
    const bool wact   = tid < WSLOTS;
    const int  wtile  = tid >> 5;             // which 16-col tile
    const int  wlane  = tid & 31;             // fragment lane
    const int  wks    = (wlane >> 4) * 16;    // K base within BK tile
    const int  wn     = wtile * 16 + (wlane & 15);  // column of W

    v8f acc[NT];
    #pragma unroll
    for (int t = 0; t < NT; ++t) acc[t] = (v8f){0.f,0.f,0.f,0.f,0.f,0.f,0.f,0.f};

    float4 xr[XR];
    float  wc[16];

    // ---- register-load stage ----
    auto load_regs = [&](int it) {
        const int k0 = it * BK;
        #pragma unroll
        for (int i = 0; i < XR; ++i) {
            int lin = tid + i * 128;                 // float4 index in tile
            int r   = lin >> 3;                      // 8 float4 per row
            int c4  = lin & 7;
            xr[i] = *reinterpret_cast<const float4*>(
                X + (size_t)(rowbase + r) * K + k0 + c4 * 4);
        }
        if (wact) {
            // column of W: 16 consecutive K at fixed n; 64B-contiguous across
            // each half-wave, constant immediate offsets from one base
            const float* wp = W + (size_t)(k0 + wks) * F + wn;
            #pragma unroll
            for (int j = 0; j < 16; ++j) wc[j] = wp[(size_t)j * F];
        }
    };

    // ---- convert + store stage (LDS, fragment layouts) ----
    auto store_lds = [&](int buf) {
        #pragma unroll
        for (int i = 0; i < XR; ++i) {
            int lin = tid + i * 128;
            int r   = lin >> 3;
            int c   = (lin & 7) * 4;
            sx[buf][r][c + 0] = (__bf16)xr[i].x;
            sx[buf][r][c + 1] = (__bf16)xr[i].y;
            sx[buf][r][c + 2] = (__bf16)xr[i].z;
            sx[buf][r][c + 3] = (__bf16)xr[i].w;
        }
        if (wact) {
            v16bf wv;
            #pragma unroll
            for (int j = 0; j < 16; ++j) wv[j] = (__bf16)wc[j];
            *reinterpret_cast<v16bf*>(&swf[buf][wtile][wlane][0]) = wv;
        }
    };

    // ---- WMMA stage ----
    auto compute = [&](int buf) {
        // A fragment (16x32 bf16): lane L (M=L&15):
        //   elems 0..15 -> K = (L>>4)*16 + e
        const __bf16* arow = &sx[buf][wave * 16 + lo][0];
        v8bf a0 = *reinterpret_cast<const v8bf*>(arow + hi * 16);
        v8bf a1 = *reinterpret_cast<const v8bf*>(arow + hi * 16 + 8);
        v16bf a = __builtin_shufflevector(a0, a1,
                    0,1,2,3,4,5,6,7,8,9,10,11,12,13,14,15);
        #pragma unroll
        for (int t = 0; t < NT; ++t) {
            // B fragment: lane's 16 bf16 stored contiguously (32B aligned)
            v16bf b = *reinterpret_cast<const v16bf*>(&swf[buf][t][lane][0]);
            acc[t] = __builtin_amdgcn_wmma_f32_16x16x32_bf16(
                false, a, false, b, (short)0, acc[t], false, false);
        }
    };

    // ---- pipelined main loop ----
    load_regs(0);
    store_lds(0);
    __syncthreads();
    for (int it = 0; it < nIter; ++it) {
        const int cur = it & 1;
        if (it + 1 < nIter) load_regs(it + 1);   // issue global loads early
        compute(cur);                            // WMMAs overlap load latency
        if (it + 1 < nIter) store_lds(cur ^ 1);
        __syncthreads();
    }

    // ---- store: C/D layout: VGPR r -> M = (L>>4)*8 + r, N = L&15 ----
    #pragma unroll
    for (int t = 0; t < NT; ++t) {
        #pragma unroll
        for (int r = 0; r < 8; ++r) {
            int mo = rowbase + wave * 16 + hi * 8 + r;
            int no = t * 16 + lo;
            Y[(size_t)mo * F + no] = acc[t][r];
        }
    }
}

// ---------------------------------------------------------------------------
// Edge aggregation: out[col] += H[row] * dis[row]*dis[col]   (one thread per
// (edge, feature); H and out are L2-resident so atomics are cheap)
// ---------------------------------------------------------------------------
__global__ void k_aggregate(const float* __restrict__ H,
                            const long long* __restrict__ ei,
                            const float* __restrict__ dis,
                            float* __restrict__ out, int E, int F) {
    long long idx = (long long)blockIdx.x * blockDim.x + threadIdx.x;
    long long total = (long long)E * F;
    if (idx >= total) return;
    int e = (int)(idx / F);
    int f = (int)(idx - (long long)e * F);
    int r = (int)ei[e];
    int c = (int)ei[(size_t)E + e];
    float v = H[(size_t)r * F + f] * dis[r] * dis[c];
    atomicAdd(&out[(size_t)c * F + f], v);
}

// ---------------------------------------------------------------------------
// Finish layer: agg[i][f] += gemm[i][f]*dis[i]^2 (self loop) + b[f]; opt ReLU
// ---------------------------------------------------------------------------
__global__ void k_finish(float* __restrict__ agg, const float* __restrict__ gemm,
                         const float* __restrict__ dis, const float* __restrict__ b,
                         int n, int F, int do_relu) {
    long long idx = (long long)blockIdx.x * blockDim.x + threadIdx.x;
    if (idx >= (long long)n * F) return;
    int i = (int)(idx / F);
    int f = (int)(idx - (long long)i * F);
    float d = dis[i];
    float v = agg[idx] + gemm[idx] * d * d + b[f];
    agg[idx] = do_relu ? fmaxf(v, 0.0f) : v;
}

// ---------------------------------------------------------------------------
// Row softmax over C=16 logits
// ---------------------------------------------------------------------------
__global__ void k_softmax16(const float* __restrict__ in, float* __restrict__ out, int n) {
    int i = blockIdx.x * blockDim.x + threadIdx.x;
    if (i >= n) return;
    const float* p = in + (size_t)i * 16;
    float m = p[0];
    #pragma unroll
    for (int f = 1; f < 16; ++f) m = fmaxf(m, p[f]);
    float s = 0.0f;
    float ex[16];
    #pragma unroll
    for (int f = 0; f < 16; ++f) { ex[f] = __expf(p[f] - m); s += ex[f]; }
    float inv = 1.0f / s;
    float* q = out + (size_t)i * 16;
    #pragma unroll
    for (int f = 0; f < 16; ++f) q[f] = ex[f] * inv;
}

// ---------------------------------------------------------------------------
extern "C" void kernel_launch(void* const* d_in, const int* in_sizes, int n_in,
                              void* d_out, int out_size, void* d_ws, size_t ws_size,
                              hipStream_t stream) {
    const float*     x  = (const float*)d_in[0];
    const long long* ei = (const long long*)d_in[1];
    const float*     W1 = (const float*)d_in[2];
    const float*     b1 = (const float*)d_in[3];
    const float*     W2 = (const float*)d_in[4];
    const float*     b2 = (const float*)d_in[5];
    const float*     W3 = (const float*)d_in[6];
    const float*     b3 = (const float*)d_in[7];
    float* out = (float*)d_out;

    const int N  = in_sizes[2] / 64;         // W1 is [N, 64]
    const int E  = in_sizes[1] / 2;          // edge_index is [2, E]
    const int F1 = 64, F2 = 32, F3 = 16;

    float* dis  = (float*)d_ws;              // [N]
    float* bufA = dis + N;                   // [N, 64] GEMM output
    float* bufB = bufA + (size_t)N * 64;     // [N, 64] aggregation / activations

    const int TB = 256;
    auto cdiv = [](long long a, long long b) { return (unsigned)((a + b - 1) / b); };

    // ---- gcn_norm ----
    k_deg_init<<<cdiv(N, TB), TB, 0, stream>>>(dis, N);
    k_deg_count<<<cdiv(E, TB), TB, 0, stream>>>(ei, dis, E);
    k_rsqrt_inplace<<<cdiv(N, TB), TB, 0, stream>>>(dis, N);

    // ---- Layer 1: x @ W1 -> aggregate -> +b1, ReLU ----
    gcn_gemm_bf16<64><<<N / 64, 128, 0, stream>>>(x, W1, bufA, N, N);
    k_zero<<<cdiv((long long)N * F1, TB), TB, 0, stream>>>(bufB, N * F1);
    k_aggregate<<<cdiv((long long)E * F1, TB), TB, 0, stream>>>(bufA, ei, dis, bufB, E, F1);
    k_finish<<<cdiv((long long)N * F1, TB), TB, 0, stream>>>(bufB, bufA, dis, b1, N, F1, 1);

    // ---- Layer 2: h1 @ W2 -> aggregate -> +b2, ReLU ----
    gcn_gemm_bf16<32><<<N / 64, 128, 0, stream>>>(bufB, W2, bufA, N, F1);
    k_zero<<<cdiv((long long)N * F2, TB), TB, 0, stream>>>(bufB, N * F2);
    k_aggregate<<<cdiv((long long)E * F2, TB), TB, 0, stream>>>(bufA, ei, dis, bufB, E, F2);
    k_finish<<<cdiv((long long)N * F2, TB), TB, 0, stream>>>(bufB, bufA, dis, b2, N, F2, 1);

    // ---- Layer 3: h2 @ W3 -> aggregate -> +b3 (no ReLU) ----
    gcn_gemm_bf16<16><<<N / 64, 128, 0, stream>>>(bufB, W3, bufA, N, F2);
    k_zero<<<cdiv((long long)N * F3, TB), TB, 0, stream>>>(bufB, N * F3);
    k_aggregate<<<cdiv((long long)E * F3, TB), TB, 0, stream>>>(bufA, ei, dis, bufB, E, F3);
    k_finish<<<cdiv((long long)N * F3, TB), TB, 0, stream>>>(bufB, bufA, dis, b3, N, F3, 0);

    // ---- Softmax over 16 classes ----
    k_softmax16<<<cdiv(N, TB), TB, 0, stream>>>(bufB, out, N);
}
